// GNNLayer_67070209294346
// MI455X (gfx1250) — matrix operand order repaired and checked by reading, not compiled, not run on hardware
//
#include <hip/hip_runtime.h>
#include <hip/hip_bf16.h>

typedef __attribute__((ext_vector_type(16))) __bf16 v16bf;
typedef __attribute__((ext_vector_type(8)))  float  v8f;

#define N_NODES   100000
#define E_EDGES   1600000
#define DIN       512
#define DOUT      64
#define NEG_SLOPE 0.2f

// ---------------------------------------------------------------------------
// Zero a float buffer (d_out is poisoned by the harness).
// ---------------------------------------------------------------------------
__global__ void zero_f32(float* __restrict__ p, int n) {
    int i = blockIdx.x * blockDim.x + threadIdx.x;
    if (i < n) p[i] = 0.0f;
}

// ---------------------------------------------------------------------------
// Repack weight [512,64] fp32 -> bf16 in WMMA B-fragment order so the GEMM
// can fetch each lane's 16 K-values with two contiguous b128 loads.
// Packed index: ((ntile*16 + kstep)*32 + lane)*16 + i
//   n = ntile*16 + (lane & 15)
//   k = kstep*32 + (lane >> 4)*8 + (i & 7) + (i >> 3)*16
//                                         (B 32x16: lanes 0-15 K=0..7,16..23,
//                                          lanes 16-31 K=8..15,24..31 per ISA layout)
// ---------------------------------------------------------------------------
__global__ void repack_weight(const float* __restrict__ w, __bf16* __restrict__ wp) {
    int t = blockIdx.x * blockDim.x + threadIdx.x;   // 0 .. 32767
    int i    = t & 15;
    int lane = (t >> 4) & 31;
    int ks   = (t >> 9) & 15;
    int nt   = t >> 13;                              // 0..3
    int n = nt * 16 + (lane & 15);
    int k = ks * 32 + (lane >> 4) * 8 + (i & 7) + (i >> 3) * 16;
    wp[t] = (__bf16)w[k * DOUT + n];
}

// ---------------------------------------------------------------------------
// support = LeakyReLU(features @ weight), bf16 WMMA with fp32 accumulate.
// Grid: 6250 blocks (16 rows each) x 128 threads (4 waves; wave = N-tile).
// ---------------------------------------------------------------------------
__global__ void __launch_bounds__(128)
gemm_lrelu(const float* __restrict__ feat,
           const __bf16* __restrict__ wp,
           float* __restrict__ support) {
    const int wave = threadIdx.x >> 5;     // N-tile 0..3
    const int lane = threadIdx.x & 31;
    const int row0 = blockIdx.x * 16;
    const int m    = lane & 15;            // A row within tile
    const int kh   = lane >> 4;            // A K-half select

    v8f acc = {};

    // A: lane holds row (row0+m); K elems: kh*8+[0..7] then 16+kh*8+[0..7]
    const float*  arow = feat + (size_t)(row0 + m) * DIN + kh * 8;
    // B: packed fragments for this wave's N-tile
    const __bf16* bp   = wp + ((size_t)wave * 16 * 32 + lane) * 16;

#pragma unroll 4
    for (int ks = 0; ks < 16; ++ks) {
        const float4* p = (const float4*)(arow + ks * 32);
        float4 f0 = p[0];   // K = kh*8 + 0..3
        float4 f1 = p[1];   // K = kh*8 + 4..7
        float4 f2 = p[4];   // K = 16 + kh*8 + 0..3
        float4 f3 = p[5];   // K = 16 + kh*8 + 4..7

        v16bf a;
        a[0]  = (__bf16)f0.x; a[1]  = (__bf16)f0.y;
        a[2]  = (__bf16)f0.z; a[3]  = (__bf16)f0.w;
        a[4]  = (__bf16)f1.x; a[5]  = (__bf16)f1.y;
        a[6]  = (__bf16)f1.z; a[7]  = (__bf16)f1.w;
        a[8]  = (__bf16)f2.x; a[9]  = (__bf16)f2.y;
        a[10] = (__bf16)f2.z; a[11] = (__bf16)f2.w;
        a[12] = (__bf16)f3.x; a[13] = (__bf16)f3.y;
        a[14] = (__bf16)f3.z; a[15] = (__bf16)f3.w;

        v16bf b = *(const v16bf*)(bp + (size_t)ks * 32 * 16);

        acc = __builtin_amdgcn_wmma_f32_16x16x32_bf16(
                  false, a, false, b, (short)0, acc, false, false);
    }

    // D layout: vgpr r -> row = row0 + kh*8 + r, col = wave*16 + m
    const int col = wave * 16 + m;
#pragma unroll
    for (int r = 0; r < 8; ++r) {
        float v = acc[r];
        v = (v > 0.0f) ? v : NEG_SLOPE * v;
        support[(size_t)(row0 + kh * 8 + r) * DOUT + col] = v;
    }
}

// ---------------------------------------------------------------------------
// out[rows[e], :] += vals[e] * x[cols[e], :]   (one wave per edge, 2 cols/lane)
// ---------------------------------------------------------------------------
__global__ void __launch_bounds__(256)
spmm_scatter(const int* __restrict__ rows,
             const int* __restrict__ cols,
             const float* __restrict__ vals,
             const float* __restrict__ x,
             float* __restrict__ out,
             int E) {
    int e = blockIdx.x * 8 + (threadIdx.x >> 5);
    if (e >= E) return;
    int   lane = threadIdx.x & 31;
    int   r = rows[e];
    int   c = cols[e];
    float v = vals[e];

    float2 xv = *(const float2*)(x + (size_t)c * DOUT + lane * 2);
    float* o  = out + (size_t)r * DOUT + lane * 2;
    unsafeAtomicAdd(o,     v * xv.x);   // global_atomic_add_f32, no return
    unsafeAtomicAdd(o + 1, v * xv.y);
}

// ---------------------------------------------------------------------------
extern "C" void kernel_launch(void* const* d_in, const int* in_sizes, int n_in,
                              void* d_out, int out_size, void* d_ws, size_t ws_size,
                              hipStream_t stream) {
    const float* feat   = (const float*)d_in[0];   // [N, 512]
    const float* weight = (const float*)d_in[1];   // [512, 64]
    const int*   erows  = (const int*)  d_in[2];   // [E]
    const int*   ecols  = (const int*)  d_in[3];   // [E]
    const float* evals  = (const float*)d_in[4];   // [E]

    float* out = (float*)d_out;                        // output [N*64]
    float* az  = out + (size_t)N_NODES * DOUT;         // az     [N*64]
    __bf16* wp = (__bf16*)d_ws;                        // 64 KB packed weight

    const int NV = N_NODES * DOUT;                     // 6.4M floats per half

    // 1) zero the "output" half (accumulated by spmm1)
    zero_f32<<<(NV + 255) / 256, 256, 0, stream>>>(out, NV);

    // 2) repack weight to bf16 B-fragments
    repack_weight<<<(DIN * DOUT) / 256, 256, 0, stream>>>(weight, wp);

    // 3) support = LeakyReLU(feat @ W), staged in the az half of d_out
    gemm_lrelu<<<N_NODES / 16, 128, 0, stream>>>(feat, wp, az);

    // 4) output = A @ support
    spmm_scatter<<<(E_EDGES + 7) / 8, 256, 0, stream>>>(
        erows, ecols, evals, /*x=*/az, /*out=*/out, E_EDGES);

    // 5) reset az half, then az = A @ output
    zero_f32<<<(NV + 255) / 256, 256, 0, stream>>>(az, NV);
    spmm_scatter<<<(E_EDGES + 7) / 8, 256, 0, stream>>>(
        erows, ecols, evals, /*x=*/out, /*out=*/az, E_EDGES);
}
